// TransPlanner_59914793779395
// MI455X (gfx1250) — matrix operand order repaired and checked by reading, not compiled, not run on hardware
//
#include <hip/hip_runtime.h>
#include <hip/hip_bf16.h>

typedef __attribute__((ext_vector_type(16))) _Float16 v16h;
typedef __attribute__((ext_vector_type(8)))  _Float16 v8h;
typedef __attribute__((ext_vector_type(4)))  _Float16 v4h;
typedef __attribute__((ext_vector_type(8)))  float    v8f;
typedef __attribute__((ext_vector_type(4)))  float    v4f;

#define B_       64
#define T_       256
#define NV_      20000
#define DEG_     8
#define E_       128
#define NBUCKET_ 288
#define FDIST    50
#define FDIR     50
#define FTT      50
#define HID_     100
#define FEED_    356
#define FEED_PAD 384
#define HID_PAD  112
#define H1K_PAD  128
#define H2_PAD   64

union H16 { v16h v; v8h h8[2]; _Float16 h[16]; };
union F32x8 { v8f v; float f[8]; };

// ---- WMMA fragment helpers (ISA 7.12.2, wave32) ----
// A 16x32 frag from row-major [16 x lda]: lane's 16 elems = two contiguous 16B chunks.
__device__ __forceinline__ v16h frag_a_vec(const _Float16* p, int lda) {
  int lane = threadIdx.x & 31;
  int r = lane & 15, g = lane >> 4;
  H16 a;
  a.h8[0] = *(const v8h*)(p + r * lda + g * 16);
  a.h8[1] = *(const v8h*)(p + r * lda + g * 16 + 8);
  return a.v;
}
// B^T frag (for Q @ K^T) from row-major K [keys x ld]: lane = 32 contiguous bytes.
__device__ __forceinline__ v16h frag_bt_vec(const _Float16* p, int ld) {
  int lane = threadIdx.x & 31;
  int c = lane & 15, g = lane >> 4;
  H16 b;
  b.h8[0] = *(const v8h*)(p + c * ld + g * 16);
  b.h8[1] = *(const v8h*)(p + c * ld + g * 16 + 8);
  return b.v;
}
// B frag from pre-swizzled fragment-order tile (512 f16, lane-major).
__device__ __forceinline__ v16h frag_b_sw(const _Float16* p) {
  int lane = threadIdx.x & 31;
  H16 b;
  b.h8[0] = *(const v8h*)(p + lane * 16);
  b.h8[1] = *(const v8h*)(p + lane * 16 + 8);
  return b.v;
}
// swizzled offset of B-tile element (kl in 0..31, cl in 0..15): lane=cl+16*(kl>>4), e=kl&15
__device__ __forceinline__ int bsw_off(int kl, int cl) {
  return ((cl + ((kl >> 4) << 4)) << 4) + (kl & 15);
}
__device__ __forceinline__ v8f wmma16(v16h a, v16h b, v8f c) {
  return __builtin_amdgcn_wmma_f32_16x16x32_f16(false, a, false, b, (short)0, c, false, false);
}

// feed-column permutation: internal [hid 0..127 | dest 128..255 | dist 256..305 | dir 306..355]
__device__ __forceinline__ int feed_orig_col(int k) {
  if (k < 128) return k;            // hidden
  if (k < 256) return k - 128 + 228; // dest emb
  if (k < 306) return k - 256 + 128; // dist_f
  if (k < 356) return k - 306 + 178; // dir_f
  return -1;                         // pad
}

// ---------------- kernels ----------------

__global__ void k_init(float* out, float* stats) {
  if (threadIdx.x == 0) out[0] = 0.f;
  if (threadIdx.x < 16) stats[threadIdx.x] = 0.f;
}

// Convert + swizzle all reused B operands into fragment-order tiles.
__global__ void k_prep(const float* Wq, const float* Wk, const float* Wv, const float* Wo,
                       const float* W1, const float* W2,
                       _Float16* Wq_s, _Float16* Wk_s, _Float16* Wv_s, _Float16* Wo_s,
                       _Float16* W1_s, _Float16* W2_s) {
  int i = blockIdx.x * blockDim.x + threadIdx.x;
  int n = gridDim.x * blockDim.x;
  for (int idx = i; idx < E_ * E_; idx += n) {          // 128x128: 4 kk x 8 nt tiles
    int k = idx >> 7, c = idx & 127;
    int dst = ((k >> 5) * 8 + (c >> 4)) * 512 + bsw_off(k & 31, c & 15);
    Wq_s[dst] = (_Float16)Wq[idx];
    Wk_s[dst] = (_Float16)Wk[idx];
    Wv_s[dst] = (_Float16)Wv[idx];
    Wo_s[dst] = (_Float16)Wo[idx];
  }
  for (int idx = i; idx < FEED_PAD * HID_PAD; idx += n) { // 384x112: 12 x 7 tiles, permuted rows
    int k = idx / HID_PAD, c = idx % HID_PAD;
    int o = feed_orig_col(k);
    float val = (o >= 0 && c < HID_) ? W1[o * HID_ + c] : 0.f;
    int dst = ((k >> 5) * 7 + (c >> 4)) * 512 + bsw_off(k & 31, c & 15);
    W1_s[dst] = (_Float16)val;
  }
  for (int idx = i; idx < H1K_PAD * H2_PAD; idx += n) {   // 128x64: 4 x 4 tiles
    int k = idx / H2_PAD, c = idx % H2_PAD;
    float val = (k < HID_ && c < FTT) ? W2[k * FTT + c] : 0.f;
    int dst = ((k >> 5) * 4 + (c >> 4)) * 512 + bsw_off(k & 31, c & 15);
    W2_s[dst] = (_Float16)val;
  }
}

// One wave per 16-row tile: x_e = emb[xs]; q,k row-major f16; v swizzled B-tiles.
__global__ void __launch_bounds__(32) k_qkv(const int* xs, const float* emb,
    const _Float16* Wq_s, const _Float16* Wk_s, const _Float16* Wv_s,
    _Float16* q_h, _Float16* k_h, _Float16* v_s) {
  __shared__ __align__(16) _Float16 xe[16][E_];
  long row0 = (long)blockIdx.x * 16;
  int lane = threadIdx.x;
  for (int idx = lane; idx < 16 * 32; idx += 32) {   // 4-float chunks
    int r = idx >> 5, cc = idx & 31;
    int vid = xs[row0 + r];
    v4f t = *(const v4f*)(emb + (long)vid * E_ + cc * 4);
    v4h o = { (_Float16)t.x, (_Float16)t.y, (_Float16)t.z, (_Float16)t.w };
    *(v4h*)&xe[r][cc * 4] = o;
  }
  __syncthreads();
  v16h a[4];
#pragma unroll
  for (int kk = 0; kk < 4; ++kk) a[kk] = frag_a_vec(&xe[0][kk * 32], E_);

  const _Float16* Ws[3] = {Wq_s, Wk_s, Wv_s};
  int col16 = lane & 15, g = lane >> 4;
  for (int m = 0; m < 3; ++m) {
    for (int nt = 0; nt < 8; ++nt) {
      v8f acc = {};
#pragma unroll
      for (int kk = 0; kk < 4; ++kk)
        acc = wmma16(a[kk], frag_b_sw(Ws[m] + (kk * 8 + nt) * 512), acc);
      F32x8 cf; cf.v = acc;
      if (m == 0) {
#pragma unroll
        for (int vv = 0; vv < 8; ++vv)
          q_h[(row0 + vv + g * 8) * E_ + nt * 16 + col16] = (_Float16)cf.f[vv];
      } else if (m == 1) {
#pragma unroll
        for (int vv = 0; vv < 8; ++vv)
          k_h[(row0 + vv + g * 8) * E_ + nt * 16 + col16] = (_Float16)cf.f[vv];
      } else {
#pragma unroll
        for (int vv = 0; vv < 8; ++vv) {            // v -> swizzled B tiles per batch
          long rg = row0 + vv + g * 8;
          int b = (int)(rg >> 8), t = (int)(rg & 255);
          long tile = (long)b * 64 + (t >> 5) * 8 + nt;
          v_s[tile * 512 + bsw_off(t & 31, col16)] = (_Float16)cf.f[vv];
        }
      }
    }
  }
}

// One wave per (batch, 16-query tile): scores -> softmax -> PV -> @Wo
__global__ void __launch_bounds__(32) k_attn(const _Float16* q_h, const _Float16* k_h,
    const _Float16* v_s, const _Float16* Wo_s, const int* lengths, _Float16* hid_h) {
  __shared__ __align__(16) float    sc[16][T_];
  __shared__ __align__(16) _Float16 P[16][T_];
  __shared__ __align__(16) _Float16 ao[16][E_];
  int bid = blockIdx.x;
  int b = bid >> 4, qt = bid & 15;
  int lane = threadIdx.x;
  int len = lengths[b];
  long base = (long)b * T_;
  int col16 = lane & 15, g = lane >> 4;

  v16h aq[4];
#pragma unroll
  for (int kk = 0; kk < 4; ++kk)
    aq[kk] = frag_a_vec(q_h + (base + qt * 16) * E_ + kk * 32, E_);

  for (int kt = 0; kt < 16; ++kt) {
    v8f acc = {};
#pragma unroll
    for (int kk = 0; kk < 4; ++kk)
      acc = wmma16(aq[kk], frag_bt_vec(k_h + (base + kt * 16) * E_ + kk * 32, E_), acc);
    F32x8 cf; cf.v = acc;
#pragma unroll
    for (int vv = 0; vv < 8; ++vv) {
      int r = vv + g * 8;
      int qi = qt * 16 + r, kj = kt * 16 + col16;
      float s = cf.f[vv] * 0.08838834764831845f;   // 1/sqrt(128)
      if (kj > qi || kj >= len) s = -1e9f;
      sc[r][kj] = s;
    }
  }
  __syncthreads();
  if (lane < 16) {
    float m = -1e30f;
    for (int j = 0; j < T_; ++j) m = fmaxf(m, sc[lane][j]);
    float ssum = 0.f;
    for (int j = 0; j < T_; ++j) { float e = __expf(sc[lane][j] - m); sc[lane][j] = e; ssum += e; }
    float inv = 1.f / ssum;
    for (int j = 0; j < T_; ++j) P[lane][j] = (_Float16)(sc[lane][j] * inv);
  }
  __syncthreads();

  v16h ap[8];
#pragma unroll
  for (int kk = 0; kk < 8; ++kk) ap[kk] = frag_a_vec(&P[0][kk * 32], T_);
  for (int nt = 0; nt < 8; ++nt) {            // PV, V pre-swizzled
    v8f acc = {};
#pragma unroll
    for (int kk = 0; kk < 8; ++kk)
      acc = wmma16(ap[kk], frag_b_sw(v_s + ((long)b * 64 + kk * 8 + nt) * 512), acc);
    F32x8 cf; cf.v = acc;
#pragma unroll
    for (int vv = 0; vv < 8; ++vv) ao[vv + g * 8][nt * 16 + col16] = (_Float16)cf.f[vv];
  }
  __syncthreads();

  v16h a2[4];
#pragma unroll
  for (int kk = 0; kk < 4; ++kk) a2[kk] = frag_a_vec(&ao[0][kk * 32], E_);
  for (int nt = 0; nt < 8; ++nt) {            // @ Wo
    v8f acc = {};
#pragma unroll
    for (int kk = 0; kk < 4; ++kk)
      acc = wmma16(a2[kk], frag_b_sw(Wo_s + (kk * 8 + nt) * 512), acc);
    F32x8 cf; cf.v = acc;
#pragma unroll
    for (int vv = 0; vv < 8; ++vv)
      hid_h[(base + qt * 16 + vv + g * 8) * E_ + nt * 16 + col16] = (_Float16)cf.f[vv];
  }
}

__global__ void k_stats(const int* xs, const int* lengths, const int* adj, const int* segs,
                        const int* start_time, const float* tds, float* stats) {
  __shared__ float red[3][256];
  long i = blockIdx.x * blockDim.x + threadIdx.x;
  long stride = (long)gridDim.x * blockDim.x;
  float s1 = 0.f, s2 = 0.f, c = 0.f;
  int st = start_time[0];
  for (long idx = i; idx < (long)B_ * T_ * DEG_; idx += stride) {
    int d = idx & 7;
    long bt = idx >> 3;
    int b = (int)(bt >> 8), t = (int)(bt & 255);
    int vid = xs[bt];
    bool valid = t < lengths[b] - 1;
    int rn = valid ? adj[vid * DEG_ + d] : 0;
    float mf = (rn != 0) ? 1.f : 0.f;
    float tt = tds[(long)segs[vid * DEG_ + d] * NBUCKET_ + st];
    s1 += tt * mf; s2 += tt * tt * mf; c += mf;
  }
  red[0][threadIdx.x] = s1; red[1][threadIdx.x] = s2; red[2][threadIdx.x] = c;
  __syncthreads();
  for (int w = 128; w > 0; w >>= 1) {
    if ((int)threadIdx.x < w) {
      red[0][threadIdx.x] += red[0][threadIdx.x + w];
      red[1][threadIdx.x] += red[1][threadIdx.x + w];
      red[2][threadIdx.x] += red[2][threadIdx.x + w];
    }
    __syncthreads();
  }
  if (threadIdx.x == 0) {
    atomicAdd(&stats[0], red[0][0]);
    atomicAdd(&stats[1], red[1][0]);
    atomicAdd(&stats[2], red[2][0]);
  }
}

// One wave per 16 neighbor-rows. Internal feed layout: [hid|dest|dist|dir] (W1 rows permuted to match).
__global__ void __launch_bounds__(32) k_mlp(
    const int* xs, const int* lengths, const int* dest, const int* adj, const int* segs,
    const int* start_time, const float* loc, const float* tds, const float* emb,
    const _Float16* hid_h, const _Float16* W1_s, const _Float16* W2_s,
    const float* distW, const float* distB, const float* dirW, const float* dirB,
    const float* ttW, const float* ttB,
    const float* obW1, const float* obB1, const float* obW2, const float* obB2,
    const float* omB1, const float* omB2, const float* omW3, const float* omB3,
    const float* stats, float* logits) {
  __shared__ __align__(16) _Float16 feed[16][FEED_PAD];
  __shared__ __align__(16) _Float16 h1[16][H1K_PAD];
  __shared__ __align__(16) float    h2[16][H2_PAD];
  __shared__ float obv[16];
  int lane = threadIdx.x;
  long row0 = (long)blockIdx.x * 16;
  float cnt  = stats[2];
  float mean = stats[0] / cnt;
  float var  = (stats[1] - mean * mean * cnt) / (cnt - 1.f);
  float rdenom = 1.f / (sqrtf(var) + 1e-6f);
  int st = start_time[0];

  for (int idx = lane; idx < 16 * 16; idx += 32) {      // hidden: 8-f16 chunks
    int r = idx >> 4, cc = idx & 15;
    long bt = (row0 + r) >> 3;
    *(v8h*)&feed[r][cc * 8] = *(const v8h*)(hid_h + bt * E_ + cc * 8);
  }
  for (int idx = lane; idx < 16 * 32; idx += 32) {      // dest emb: 4-f32 chunks -> cols 128..255
    int r = idx >> 5, cc = idx & 31;
    int bb = (int)(((row0 + r) >> 3) >> 8);
    v4f t = *(const v4f*)(emb + (long)dest[bb] * E_ + cc * 4);
    v4h o = { (_Float16)t.x, (_Float16)t.y, (_Float16)t.z, (_Float16)t.w };
    *(v4h*)&feed[r][128 + cc * 4] = o;
  }
  if (lane < 16) {                                      // per-row scalar features
    long rr = row0 + lane;
    long bt = rr >> 3;
    int d = rr & 7;
    int b = (int)(bt >> 8), t = (int)(bt & 255);
    int vid = xs[bt];
    bool valid = t < lengths[b] - 1;
    int rn = valid ? adj[vid * DEG_ + d] : 0;
    float mf = (rn != 0) ? 1.f : 0.f;
    float nx = loc[rn * 2], ny = loc[rn * 2 + 1];
    int db = dest[b];
    float dx = loc[db * 2], dy = loc[db * 2 + 1];
    float dist = (fabsf(nx - dx) + fabsf(ny - dy)) * 100.f;
    float cx = loc[vid * 2], cy = loc[vid * 2 + 1];
    float vx = nx - cx, vy = ny - cy;
    float nrm = sqrtf(vx * vx + vy * vy) + 1e-8f;
    float ux = vx / nrm, uy = vy / nrm;
    for (int j = 0; j < FDIST; ++j)
      feed[lane][256 + j] = (_Float16)(dist * distW[j] + distB[j]);
    for (int j = 0; j < FDIR; ++j)
      feed[lane][306 + j] = (_Float16)(ux * dirW[j] + uy * dirW[FDIR + j] + dirB[j]);
    for (int j = FEED_; j < FEED_PAD; ++j) feed[lane][j] = (_Float16)0.f;
    for (int j = HID_PAD; j < H1K_PAD; ++j) h1[lane][j] = (_Float16)0.f;
    // ob branch (VALU)
    float tt  = tds[(long)segs[vid * DEG_ + d] * NBUCKET_ + st];
    float ttn = (tt - mean) * rdenom;
    float ob = obB2[0];
    for (int m = 0; m < 25; ++m) {
      float h = obB1[m];
      for (int j = 0; j < FTT; ++j)
        h += ((ttn * ttW[j] + ttB[j]) * mf) * obW1[j * 25 + m];
      ob += fmaxf(h, 0.f) * obW2[m];
    }
    obv[lane] = ob;
  }
  __syncthreads();

  v16h af[12];
#pragma unroll
  for (int kk = 0; kk < 12; ++kk) af[kk] = frag_a_vec(&feed[0][kk * 32], FEED_PAD);
  int col16 = lane & 15, g = lane >> 4;
  for (int nt = 0; nt < 7; ++nt) {            // h1 = relu(feed @ W1 + b1)
    v8f acc = {};
#pragma unroll
    for (int kk = 0; kk < 12; ++kk)
      acc = wmma16(af[kk], frag_b_sw(W1_s + (kk * 7 + nt) * 512), acc);
    int col = col16 + nt * 16;
    F32x8 cf; cf.v = acc;
#pragma unroll
    for (int vv = 0; vv < 8; ++vv) {
      float bb = (col < HID_) ? omB1[col] : 0.f;
      h1[vv + g * 8][col] = (_Float16)fmaxf(cf.f[vv] + bb, 0.f);
    }
  }
  __syncthreads();
  v16h ah[4];
#pragma unroll
  for (int kk = 0; kk < 4; ++kk) ah[kk] = frag_a_vec(&h1[0][kk * 32], H1K_PAD);
  for (int nt = 0; nt < 4; ++nt) {            // h2 = relu(h1 @ W2 + b2)
    v8f acc = {};
#pragma unroll
    for (int kk = 0; kk < 4; ++kk)
      acc = wmma16(ah[kk], frag_b_sw(W2_s + (kk * 4 + nt) * 512), acc);
    int col = col16 + nt * 16;
    F32x8 cf; cf.v = acc;
#pragma unroll
    for (int vv = 0; vv < 8; ++vv) {
      float bb = (col < FTT) ? omB2[col] : 0.f;
      h2[vv + g * 8][col] = (col < FTT) ? fmaxf(cf.f[vv] + bb, 0.f) : 0.f;
    }
  }
  __syncthreads();
  if (lane < 16) {                             // head: h2 @ W3 + b3 + ob
    float s = omB3[0] + obv[lane];
    for (int j = 0; j < FTT; ++j) s += h2[lane][j] * omW3[j];
    logits[row0 + lane] = s;
  }
}

__global__ void __launch_bounds__(32) k_loss(const float* logits, const int* actions,
                                             const int* lengths, float* out) {
  int b = blockIdx.x;
  int lane = threadIdx.x;
  int len = lengths[b];
  float acc = 0.f;
  for (int t = lane; t < len - 1; t += 32) {
    const float* l = logits + ((long)b * T_ + t) * DEG_;
    float m = l[0];
    for (int j = 1; j < 8; ++j) m = fmaxf(m, l[j]);
    float s = 0.f;
    for (int j = 0; j < 8; ++j) s += __expf(l[j] - m);
    int a = actions[b * (T_ - 1) + t];
    acc += -(l[a] - m - __logf(s));
  }
  for (int w = 16; w > 0; w >>= 1) acc += __shfl_xor(acc, w, 32);
  if (lane == 0) atomicAdd(out, acc / (float)(len - 1));
}

extern "C" void kernel_launch(void* const* d_in, const int* in_sizes, int n_in,
                              void* d_out, int out_size, void* d_ws, size_t ws_size,
                              hipStream_t stream) {
  (void)in_sizes; (void)n_in; (void)out_size; (void)ws_size;
  const int*   xs      = (const int*)d_in[0];
  const int*   lengths = (const int*)d_in[1];
  const int*   actions = (const int*)d_in[2];
  const int*   dest    = (const int*)d_in[3];
  const int*   adj     = (const int*)d_in[4];
  const int*   segs    = (const int*)d_in[5];
  const int*   start_t = (const int*)d_in[6];
  const float* loc     = (const float*)d_in[7];
  const float* tds     = (const float*)d_in[8];
  const float* emb     = (const float*)d_in[9];
  const float* Wq      = (const float*)d_in[10];
  const float* Wk      = (const float*)d_in[11];
  const float* Wv      = (const float*)d_in[12];
  const float* Wo      = (const float*)d_in[13];
  const float* distW   = (const float*)d_in[14];
  const float* distB   = (const float*)d_in[15];
  const float* dirW    = (const float*)d_in[16];
  const float* dirB    = (const float*)d_in[17];
  const float* ttW     = (const float*)d_in[18];
  const float* ttB     = (const float*)d_in[19];
  const float* obW1    = (const float*)d_in[20];
  const float* obB1    = (const float*)d_in[21];
  const float* obW2    = (const float*)d_in[22];
  const float* obB2    = (const float*)d_in[23];
  const float* omW1    = (const float*)d_in[24];
  const float* omB1    = (const float*)d_in[25];
  const float* omW2    = (const float*)d_in[26];
  const float* omB2    = (const float*)d_in[27];
  const float* omW3    = (const float*)d_in[28];
  const float* omB3    = (const float*)d_in[29];

  char* ws = (char*)d_ws;
  size_t off = 0;
  auto carve = [&](size_t bytes) -> char* {
    char* p = ws + off;
    off = (off + bytes + 255) & ~(size_t)255;
    return p;
  };
  _Float16* Wq_s  = (_Float16*)carve((size_t)E_ * E_ * 2);
  _Float16* Wk_s  = (_Float16*)carve((size_t)E_ * E_ * 2);
  _Float16* Wv_s  = (_Float16*)carve((size_t)E_ * E_ * 2);
  _Float16* Wo_s  = (_Float16*)carve((size_t)E_ * E_ * 2);
  _Float16* W1_s  = (_Float16*)carve((size_t)FEED_PAD * HID_PAD * 2);
  _Float16* W2_s  = (_Float16*)carve((size_t)H1K_PAD * H2_PAD * 2);
  _Float16* q_h   = (_Float16*)carve((size_t)B_ * T_ * E_ * 2);
  _Float16* k_h   = (_Float16*)carve((size_t)B_ * T_ * E_ * 2);
  _Float16* v_s   = (_Float16*)carve((size_t)B_ * T_ * E_ * 2);
  _Float16* hid_h = (_Float16*)carve((size_t)B_ * T_ * E_ * 2);
  float*    logits= (float*)   carve((size_t)B_ * T_ * DEG_ * 4);
  float*    stats = (float*)   carve(16 * 4);

  k_init<<<1, 32, 0, stream>>>((float*)d_out, stats);
  k_prep<<<256, 256, 0, stream>>>(Wq, Wk, Wv, Wo, omW1, omW2,
                                  Wq_s, Wk_s, Wv_s, Wo_s, W1_s, W2_s);
  k_qkv<<<(B_ * T_) / 16, 32, 0, stream>>>(xs, emb, Wq_s, Wk_s, Wv_s, q_h, k_h, v_s);
  k_attn<<<B_ * (T_ / 16), 32, 0, stream>>>(q_h, k_h, v_s, Wo_s, lengths, hid_h);
  k_stats<<<256, 256, 0, stream>>>(xs, lengths, adj, segs, start_t, tds, stats);
  k_mlp<<<(B_ * T_ * DEG_) / 16, 32, 0, stream>>>(
      xs, lengths, dest, adj, segs, start_t, loc, tds, emb,
      hid_h, W1_s, W2_s, distW, distB, dirW, dirB, ttW, ttB,
      obW1, obB1, obW2, obB2, omB1, omB2, omW3, omB3, stats, logits);
  k_loss<<<B_, 32, 0, stream>>>(logits, actions, lengths, (float*)d_out);
}
